// Combined_model_53360673685573
// MI455X (gfx1250) — matrix operand order repaired and checked by reading, not compiled, not run on hardware
//
#include <hip/hip_runtime.h>
#include <hip/hip_bf16.h>
#include <stdint.h>

// ---------------- problem constants (match reference) ----------------
#define NN     50000
#define EE     800000
#define FF     64
#define HH     4
#define CC     16
#define GG     512
#define DBIN   1024
#define DB1    512
#define DB2    256
#define DB3    128
#define DBOUT  64
#define DA1    128
#define DC     32
#define BN_EPS 1e-5f
#define E2     (EE + NN)   // edges + self loops

typedef __attribute__((ext_vector_type(16))) _Float16 v16h;
typedef __attribute__((ext_vector_type(8)))  float    v8f;

// ---------------- epilogue modes ----------------
#define EP_RAW       0   // y = acc
#define EP_BIAS_RELU 1   // y = relu(acc + bias)
#define EP_BN_RELU   2   // y = relu((acc + bias)*scale + shift)
#define EP_BIAS_SIG  3   // y = sigmoid(acc + bias)

__device__ __forceinline__ float sigmoidf_(float x) { return 1.0f / (1.0f + __expf(-x)); }

// CDNA5 async copy: global -> LDS with per-lane LDS destination (ASYNCcnt).
__device__ __forceinline__ void async_g2l_b32(unsigned lds_off, const float* gptr) {
  asm volatile("global_load_async_to_lds_b32 %0, %1, off"
               :: "v"(lds_off), "v"(gptr) : "memory");
}
__device__ __forceinline__ void wait_asynccnt0() {
  asm volatile("s_wait_asynccnt 0x0" ::: "memory");
}

// ---------------- WMMA GEMM: C[M,N] = epilogue(A[M,K] @ B[K,N]) ----------------
// fp32 in/out, f16 multiply on v_wmma_f32_16x16x32_f16.
// Block = 256 threads = 8 waves covering 8 consecutive 16-row M-tiles of one
// 16-col N-tile. The 32x16 B k-chunk is staged TRANSPOSED into LDS via
// global_load_async_to_lds_b32 (per-lane LDS scatter), shared by all 8 waves.
// Tail waves CLAMP to the last valid M-tile (duplicate compute, identical
// stores) so EXEC is all-1s for every WMMA — no divergent control flow.
// M,N multiples of 16; K multiple of 32.
#define BT_PITCH 36   // padded floats per transposed B row (16B aligned, bank-spread)

template <int EP>
__global__ void wmma_gemm_f32(const float* __restrict__ A, const float* __restrict__ B,
                              const float* __restrict__ bias,
                              const float* __restrict__ scale, const float* __restrict__ shift,
                              float* __restrict__ C, int M, int Nc, int K, int nMB) {
  __shared__ float tileBT[16 * BT_PITCH];    // [n][k] transposed B chunk

  const int tid  = threadIdx.x;
  const int wid  = tid >> 5;
  const int lane = tid & 31;
  const int half = lane >> 4;                // lane half selects K sub-range
  const int l15  = lane & 15;

  const int mblk   = blockIdx.x % nMB;
  const int tn     = (blockIdx.x / nMB) << 4;
  const int mtiles = M >> 4;
  int tileM = mblk * 8 + wid;
  if (tileM > mtiles - 1) tileM = mtiles - 1;   // clamp: EXEC stays all-1s
  const int tm = tileM << 4;

  // staging: thread handles elements tid and tid+256 of the 32x16 chunk
  const int r0 = tid >> 4, c0 = tid & 15;
  const int r1 = r0 + 16;
  const unsigned ldsBase = (unsigned)(uintptr_t)(void*)tileBT;
  const unsigned lds0 = ldsBase + (unsigned)(c0 * BT_PITCH + r0) * 4u;
  const unsigned lds1 = ldsBase + (unsigned)(c0 * BT_PITCH + r1) * 4u;

  const float* arowBase = A + (size_t)(tm + l15) * K + half * 8;
  const float* browLds  = &tileBT[l15 * BT_PITCH + half * 8];

  v8f acc = {};
  for (int kk = 0; kk < K; kk += 32) {
    // ---- cooperative async stage of B[kk..kk+32) x [tn..tn+16) (transposed) ----
    const float* g0 = B + (size_t)(kk + r0) * Nc + (tn + c0);
    const float* g1 = B + (size_t)(kk + r1) * Nc + (tn + c0);
    async_g2l_b32(lds0, g0);
    async_g2l_b32(lds1, g1);
    wait_asynccnt0();
    __syncthreads();

    // A fragment: lane holds row tm+l15; two contiguous 8-float runs
    const float* arow = arowBase + kk;
    float4 a0 = *(const float4*)(arow + 0);
    float4 a1 = *(const float4*)(arow + 4);
    float4 a2 = *(const float4*)(arow + 16);
    float4 a3 = *(const float4*)(arow + 20);
    v16h af;
    af[0]  = (_Float16)a0.x; af[1]  = (_Float16)a0.y; af[2]  = (_Float16)a0.z; af[3]  = (_Float16)a0.w;
    af[4]  = (_Float16)a1.x; af[5]  = (_Float16)a1.y; af[6]  = (_Float16)a1.z; af[7]  = (_Float16)a1.w;
    af[8]  = (_Float16)a2.x; af[9]  = (_Float16)a2.y; af[10] = (_Float16)a2.z; af[11] = (_Float16)a2.w;
    af[12] = (_Float16)a3.x; af[13] = (_Float16)a3.y; af[14] = (_Float16)a3.z; af[15] = (_Float16)a3.w;

    // B fragment: lane holds col tn+l15; two 8-float runs at K = kk + 8*half and kk + 16 + 8*half
    float4 b0 = *(const float4*)(browLds + 0);
    float4 b1 = *(const float4*)(browLds + 4);
    float4 b2 = *(const float4*)(browLds + 16);
    float4 b3 = *(const float4*)(browLds + 20);
    v16h bf;
    bf[0]  = (_Float16)b0.x; bf[1]  = (_Float16)b0.y; bf[2]  = (_Float16)b0.z; bf[3]  = (_Float16)b0.w;
    bf[4]  = (_Float16)b1.x; bf[5]  = (_Float16)b1.y; bf[6]  = (_Float16)b1.z; bf[7]  = (_Float16)b1.w;
    bf[8]  = (_Float16)b2.x; bf[9]  = (_Float16)b2.y; bf[10] = (_Float16)b2.z; bf[11] = (_Float16)b2.w;
    bf[12] = (_Float16)b3.x; bf[13] = (_Float16)b3.y; bf[14] = (_Float16)b3.z; bf[15] = (_Float16)b3.w;

    acc = __builtin_amdgcn_wmma_f32_16x16x32_f16(false, af, false, bf,
                                                 (short)0, acc, false, false);
    __syncthreads();
  }

  // C/D layout: VGPR i, lanes 0-15 -> row tm+i, lanes 16-31 -> row tm+i+8
  const int col = tn + l15;
  float* cp = C + (size_t)(tm + (half << 3)) * Nc + col;
#pragma unroll
  for (int i = 0; i < 8; ++i) {
    float y = acc[i];
    if (EP == EP_BIAS_RELU)      { y += bias[col]; y = fmaxf(y, 0.0f); }
    else if (EP == EP_BN_RELU)   { y += bias[col]; y = y * scale[col] + shift[col]; y = fmaxf(y, 0.0f); }
    else if (EP == EP_BIAS_SIG)  { y += bias[col]; y = sigmoidf_(y); }
    cp[(size_t)i * Nc] = y;
  }
}

// ---------------- small helpers ----------------
__global__ void k_fill(float* __restrict__ p, float v, int n) {
  int i = blockIdx.x * blockDim.x + threadIdx.x;
  if (i < n) p[i] = v;
}

__global__ void k_bn_prep(const float* __restrict__ g, const float* __restrict__ b,
                          const float* __restrict__ m, const float* __restrict__ v,
                          float* __restrict__ scale, float* __restrict__ shift, int n) {
  int i = blockIdx.x * blockDim.x + threadIdx.x;
  if (i < n) {
    float s = g[i] * rsqrtf(v[i] + BN_EPS);
    scale[i] = s;
    shift[i] = b[i] - m[i] * s;
  }
}

// per (node, head): attention logit halves  a_s = <h[n,h,:], asrc[h,:]>
__global__ void k_att_coeff(const float* __restrict__ hA, const float* __restrict__ asrc,
                            const float* __restrict__ adst,
                            float* __restrict__ as_, float* __restrict__ ad_) {
  int t = blockIdx.x * blockDim.x + threadIdx.x;
  if (t >= NN * HH) return;
  int n = t >> 2, h = t & 3;
  const float* hp = hA + (size_t)n * (HH * CC) + h * CC;
  const float* ws = asrc + h * CC;
  const float* wd = adst + h * CC;
  float ss = 0.f, sd = 0.f;
#pragma unroll
  for (int c = 0; c < CC; ++c) { ss += hp[c] * ws[c]; sd += hp[c] * wd[c]; }
  as_[t] = ss;
  ad_[t] = sd;
}

__device__ __forceinline__ void atomicMaxF32(float* addr, float val) {
  unsigned* ua = (unsigned*)addr;
  unsigned old = *ua;
  while (true) {
    float cur = __uint_as_float(old);
    if (cur >= val) return;
    unsigned prev = atomicCAS(ua, old, __float_as_uint(val));
    if (prev == old) return;
    old = prev;
  }
}

__device__ __forceinline__ void edge_sd(int e, const int* __restrict__ ei, int& s, int& d) {
  if (e < EE) { s = ei[e]; d = ei[EE + e]; }
  else        { s = e - EE; d = e - EE; }     // implicit self loops
}

__device__ __forceinline__ float leaky02(float x) { return x > 0.f ? x : 0.2f * x; }

// pass 1: segment max of edge logits over dst
__global__ void k_edge_max(const int* __restrict__ ei, const float* __restrict__ as_,
                           const float* __restrict__ ad_, float* __restrict__ mx) {
  int t = blockIdx.x * blockDim.x + threadIdx.x;
  if (t >= E2 * HH) return;
  int e = t >> 2, h = t & 3, s, d;
  edge_sd(e, ei, s, d);
  float x = leaky02(as_[s * HH + h] + ad_[d * HH + h]);
  atomicMaxF32(&mx[d * HH + h], x);
}

// pass 2: softmax denominator
__global__ void k_edge_den(const int* __restrict__ ei, const float* __restrict__ as_,
                           const float* __restrict__ ad_, const float* __restrict__ mx,
                           float* __restrict__ den) {
  int t = blockIdx.x * blockDim.x + threadIdx.x;
  if (t >= E2 * HH) return;
  int e = t >> 2, h = t & 3, s, d;
  edge_sd(e, ei, s, d);
  float x = leaky02(as_[s * HH + h] + ad_[d * HH + h]);
  atomicAdd(&den[d * HH + h], __expf(x - mx[d * HH + h]));
}

// pass 3: weighted aggregation  out[d,h,:] += alpha * h[s,h,:]
__global__ void k_edge_agg(const int* __restrict__ ei, const float* __restrict__ as_,
                           const float* __restrict__ ad_, const float* __restrict__ mx,
                           const float* __restrict__ den, const float* __restrict__ hA,
                           float* __restrict__ out) {
  int t = blockIdx.x * blockDim.x + threadIdx.x;
  if (t >= E2 * HH) return;
  int e = t >> 2, h = t & 3, s, d;
  edge_sd(e, ei, s, d);
  const float* hs = hA + (size_t)s * (HH * CC) + h * CC;
  __builtin_prefetch(hs, 0, 1);              // global_prefetch_b8: L2-resident row
  float x = leaky02(as_[s * HH + h] + ad_[d * HH + h]);
  float alpha = __expf(x - mx[d * HH + h]) / (den[d * HH + h] + 1e-16f);
  float* od = out + (size_t)d * (HH * CC) + h * CC;
#pragma unroll
  for (int c = 0; c < CC; ++c) atomicAdd(&od[c], hs[c] * alpha);
}

// node epilogue: y = bn(relu(agg + bias))
__global__ void k_node_post(const float* __restrict__ agg, const float* __restrict__ bias,
                            const float* __restrict__ scale, const float* __restrict__ shift,
                            float* __restrict__ y) {
  int t = blockIdx.x * blockDim.x + threadIdx.x;
  if (t >= NN * FF) return;
  int c = t & (FF - 1);
  float v = agg[t] + bias[c];
  v = fmaxf(v, 0.0f);
  y[t] = v * scale[c] + shift[c];
}

// mean-pool: sums and counts
__global__ void k_pool_sum(const float* __restrict__ h, const int* __restrict__ batch,
                           float* __restrict__ pooled, float* __restrict__ cnt) {
  int t = blockIdx.x * blockDim.x + threadIdx.x;
  if (t >= NN * FF) return;
  int n = t >> 6, c = t & 63;
  int b = batch[n];
  atomicAdd(&pooled[(size_t)b * FF + c], h[t]);
  if (c == 0) atomicAdd(&cnt[b], 1.0f);
}

__global__ void k_pool_div(float* __restrict__ pooled, const float* __restrict__ cnt) {
  int t = blockIdx.x * blockDim.x + threadIdx.x;
  if (t >= GG * FF) return;
  pooled[t] /= fmaxf(cnt[t >> 6], 1.0f);
}

// modelA head layer 2: xa[g] = sigmoid(<za[g,:], w> + b)
__global__ void k_head_a2(const float* __restrict__ za, const float* __restrict__ w,
                          const float* __restrict__ b, float* __restrict__ xa) {
  int g = blockIdx.x * blockDim.x + threadIdx.x;
  if (g >= GG) return;
  float acc = b[0];
  for (int k = 0; k < DA1; ++k) acc += za[(size_t)g * DA1 + k] * w[k];
  xa[g] = sigmoidf_(acc);
}

// combined head layer 1 (K = 65 = [xa | xb]): cb[g,j] = relu(...)
__global__ void k_comb1(const float* __restrict__ xa, const float* __restrict__ xb,
                        const float* __restrict__ w, const float* __restrict__ b,
                        float* __restrict__ cb) {
  int t = blockIdx.x * blockDim.x + threadIdx.x;
  if (t >= GG * DC) return;
  int g = t / DC, j = t % DC;
  float acc = b[j] + xa[g] * w[j];                 // row 0 of [65, 32]
  const float* xr = xb + (size_t)g * DBOUT;
  for (int k = 0; k < DBOUT; ++k) acc += xr[k] * w[(size_t)(1 + k) * DC + j];
  cb[t] = fmaxf(acc, 0.0f);
}

// combined head layer 2: out[g] = sigmoid(<cb[g,:], w> + b)
__global__ void k_comb2(const float* __restrict__ cb, const float* __restrict__ w,
                        const float* __restrict__ b, float* __restrict__ out) {
  int g = blockIdx.x * blockDim.x + threadIdx.x;
  if (g >= GG) return;
  float acc = b[0];
#pragma unroll
  for (int j = 0; j < DC; ++j) acc += cb[(size_t)g * DC + j] * w[j];
  out[g] = sigmoidf_(acc);
}

// ---------------- host side ----------------
static inline int cdiv(long long a, long long b) { return (int)((a + b - 1) / b); }

static void launch_gemm(int ep, const float* A, const float* B, const float* bias,
                        const float* scale, const float* shift, float* C,
                        int M, int Nc, int K, hipStream_t s) {
  int nMB = cdiv(M >> 4, 8);                   // 8 M-tiles (8 waves) per block
  int blocks = nMB * (Nc >> 4);
  dim3 grid(blocks), blk(256);
  switch (ep) {
    case EP_RAW:       hipLaunchKernelGGL((wmma_gemm_f32<EP_RAW>),       grid, blk, 0, s, A, B, bias, scale, shift, C, M, Nc, K, nMB); break;
    case EP_BIAS_RELU: hipLaunchKernelGGL((wmma_gemm_f32<EP_BIAS_RELU>), grid, blk, 0, s, A, B, bias, scale, shift, C, M, Nc, K, nMB); break;
    case EP_BN_RELU:   hipLaunchKernelGGL((wmma_gemm_f32<EP_BN_RELU>),   grid, blk, 0, s, A, B, bias, scale, shift, C, M, Nc, K, nMB); break;
    default:           hipLaunchKernelGGL((wmma_gemm_f32<EP_BIAS_SIG>),  grid, blk, 0, s, A, B, bias, scale, shift, C, M, Nc, K, nMB); break;
  }
}

extern "C" void kernel_launch(void* const* d_in, const int* in_sizes, int n_in,
                              void* d_out, int out_size, void* d_ws, size_t ws_size,
                              hipStream_t stream) {
  (void)in_sizes; (void)n_in; (void)out_size; (void)ws_size;
  const float* x1    = (const float*)d_in[0];
  const float* x2    = (const float*)d_in[1];
  const float* gW1   = (const float*)d_in[2];
  const float* gb1   = (const float*)d_in[3];
  const float* asrc1 = (const float*)d_in[4];
  const float* adst1 = (const float*)d_in[5];
  const float* bn1_g = (const float*)d_in[6];
  const float* bn1_b = (const float*)d_in[7];
  const float* bn1_m = (const float*)d_in[8];
  const float* bn1_v = (const float*)d_in[9];
  const float* gW2   = (const float*)d_in[10];
  const float* gb2   = (const float*)d_in[11];
  const float* asrc2 = (const float*)d_in[12];
  const float* adst2 = (const float*)d_in[13];
  const float* bn2_g = (const float*)d_in[14];
  const float* bn2_b = (const float*)d_in[15];
  const float* bn2_m = (const float*)d_in[16];
  const float* bn2_v = (const float*)d_in[17];
  const float* la1_w = (const float*)d_in[18];
  const float* la1_b = (const float*)d_in[19];
  const float* la2_w = (const float*)d_in[20];
  const float* la2_b = (const float*)d_in[21];
  const float* lb1_w = (const float*)d_in[22];
  const float* lb1_b = (const float*)d_in[23];
  const float* bnb1_g = (const float*)d_in[24];
  const float* bnb1_b = (const float*)d_in[25];
  const float* bnb1_m = (const float*)d_in[26];
  const float* bnb1_v = (const float*)d_in[27];
  const float* lb2_w = (const float*)d_in[28];
  const float* lb2_b = (const float*)d_in[29];
  const float* bnb2_g = (const float*)d_in[30];
  const float* bnb2_b = (const float*)d_in[31];
  const float* bnb2_m = (const float*)d_in[32];
  const float* bnb2_v = (const float*)d_in[33];
  const float* lb3_w = (const float*)d_in[34];
  const float* lb3_b = (const float*)d_in[35];
  const float* bnb3_g = (const float*)d_in[36];
  const float* bnb3_b = (const float*)d_in[37];
  const float* bnb3_m = (const float*)d_in[38];
  const float* bnb3_v = (const float*)d_in[39];
  const float* lb4_w = (const float*)d_in[40];
  const float* lb4_b = (const float*)d_in[41];
  const float* lc1_w = (const float*)d_in[42];
  const float* lc1_b = (const float*)d_in[43];
  const float* lc2_w = (const float*)d_in[44];
  const float* lc2_b = (const float*)d_in[45];
  const int*   eidx  = (const int*)d_in[46];
  const int*   batch = (const int*)d_in[47];
  float* out = (float*)d_out;

  // -------- workspace layout (floats) --------
  float* w = (float*)d_ws;
  size_t o = 0;
  float* hA    = w + o; o += (size_t)NN * FF;     // GEMM output (per-node transformed feats)
  float* hAgg  = w + o; o += (size_t)NN * FF;     // attention-aggregated feats
  float* hPost = w + o; o += (size_t)NN * FF;     // post bias+relu+bn feats
  float* as_   = w + o; o += (size_t)NN * HH;
  float* ad_   = w + o; o += (size_t)NN * HH;
  float* mx    = w + o; o += (size_t)NN * HH;
  float* den   = w + o; o += (size_t)NN * HH;
  float* pooled= w + o; o += (size_t)GG * FF;
  float* cnt   = w + o; o += (size_t)GG;
  float* za    = w + o; o += (size_t)GG * DA1;
  float* xa    = w + o; o += (size_t)GG;
  float* zb1   = w + o; o += (size_t)GG * DB1;
  float* zb2   = w + o; o += (size_t)GG * DB2;
  float* zb3   = w + o; o += (size_t)GG * DB3;
  float* xb    = w + o; o += (size_t)GG * DBOUT;
  float* cb    = w + o; o += (size_t)GG * DC;
  float* sc1   = w + o; o += FF;   float* sh1  = w + o; o += FF;
  float* sc2   = w + o; o += FF;   float* sh2  = w + o; o += FF;
  float* scb1  = w + o; o += DB1;  float* shb1 = w + o; o += DB1;
  float* scb2  = w + o; o += DB2;  float* shb2 = w + o; o += DB2;
  float* scb3  = w + o; o += DB3;  float* shb3 = w + o; o += DB3;

  const int T = 256;
  const int nNH = NN * HH, nNF = NN * FF, nEH = E2 * HH;

  // -------- BN parameter folding --------
  hipLaunchKernelGGL(k_bn_prep, dim3(cdiv(FF, T)),  dim3(T), 0, stream, bn1_g, bn1_b, bn1_m, bn1_v, sc1, sh1, FF);
  hipLaunchKernelGGL(k_bn_prep, dim3(cdiv(FF, T)),  dim3(T), 0, stream, bn2_g, bn2_b, bn2_m, bn2_v, sc2, sh2, FF);
  hipLaunchKernelGGL(k_bn_prep, dim3(cdiv(DB1, T)), dim3(T), 0, stream, bnb1_g, bnb1_b, bnb1_m, bnb1_v, scb1, shb1, DB1);
  hipLaunchKernelGGL(k_bn_prep, dim3(cdiv(DB2, T)), dim3(T), 0, stream, bnb2_g, bnb2_b, bnb2_m, bnb2_v, scb2, shb2, DB2);
  hipLaunchKernelGGL(k_bn_prep, dim3(cdiv(DB3, T)), dim3(T), 0, stream, bnb3_g, bnb3_b, bnb3_m, bnb3_v, scb3, shb3, DB3);

  // ================= GAT layer 1 =================
  launch_gemm(EP_RAW, x1, gW1, nullptr, nullptr, nullptr, hA, NN, FF, FF, stream);
  hipLaunchKernelGGL(k_att_coeff, dim3(cdiv(nNH, T)), dim3(T), 0, stream, hA, asrc1, adst1, as_, ad_);
  hipLaunchKernelGGL(k_fill, dim3(cdiv(nNH, T)), dim3(T), 0, stream, mx, -3.0e38f, nNH);
  hipLaunchKernelGGL(k_fill, dim3(cdiv(nNH, T)), dim3(T), 0, stream, den, 0.0f, nNH);
  hipLaunchKernelGGL(k_fill, dim3(cdiv(nNF, T)), dim3(T), 0, stream, hAgg, 0.0f, nNF);
  hipLaunchKernelGGL(k_edge_max, dim3(cdiv(nEH, T)), dim3(T), 0, stream, eidx, as_, ad_, mx);
  hipLaunchKernelGGL(k_edge_den, dim3(cdiv(nEH, T)), dim3(T), 0, stream, eidx, as_, ad_, mx, den);
  hipLaunchKernelGGL(k_edge_agg, dim3(cdiv(nEH, T)), dim3(T), 0, stream, eidx, as_, ad_, mx, den, hA, hAgg);
  hipLaunchKernelGGL(k_node_post, dim3(cdiv(nNF, T)), dim3(T), 0, stream, hAgg, gb1, sc1, sh1, hPost);

  // ================= GAT layer 2 =================
  launch_gemm(EP_RAW, hPost, gW2, nullptr, nullptr, nullptr, hA, NN, FF, FF, stream);
  hipLaunchKernelGGL(k_att_coeff, dim3(cdiv(nNH, T)), dim3(T), 0, stream, hA, asrc2, adst2, as_, ad_);
  hipLaunchKernelGGL(k_fill, dim3(cdiv(nNH, T)), dim3(T), 0, stream, mx, -3.0e38f, nNH);
  hipLaunchKernelGGL(k_fill, dim3(cdiv(nNH, T)), dim3(T), 0, stream, den, 0.0f, nNH);
  hipLaunchKernelGGL(k_fill, dim3(cdiv(nNF, T)), dim3(T), 0, stream, hAgg, 0.0f, nNF);
  hipLaunchKernelGGL(k_edge_max, dim3(cdiv(nEH, T)), dim3(T), 0, stream, eidx, as_, ad_, mx);
  hipLaunchKernelGGL(k_edge_den, dim3(cdiv(nEH, T)), dim3(T), 0, stream, eidx, as_, ad_, mx, den);
  hipLaunchKernelGGL(k_edge_agg, dim3(cdiv(nEH, T)), dim3(T), 0, stream, eidx, as_, ad_, mx, den, hA, hAgg);
  hipLaunchKernelGGL(k_node_post, dim3(cdiv(nNF, T)), dim3(T), 0, stream, hAgg, gb2, sc2, sh2, hPost);

  // ================= mean pool + modelA head =================
  hipLaunchKernelGGL(k_fill, dim3(cdiv(GG * FF, T)), dim3(T), 0, stream, pooled, 0.0f, GG * FF);
  hipLaunchKernelGGL(k_fill, dim3(cdiv(GG, T)), dim3(T), 0, stream, cnt, 0.0f, GG);
  hipLaunchKernelGGL(k_pool_sum, dim3(cdiv(nNF, T)), dim3(T), 0, stream, hPost, batch, pooled, cnt);
  hipLaunchKernelGGL(k_pool_div, dim3(cdiv(GG * FF, T)), dim3(T), 0, stream, pooled, cnt);
  launch_gemm(EP_BIAS_RELU, pooled, la1_w, la1_b, nullptr, nullptr, za, GG, DA1, FF, stream);
  hipLaunchKernelGGL(k_head_a2, dim3(cdiv(GG, T)), dim3(T), 0, stream, za, la2_w, la2_b, xa);

  // ================= modelB MLP (all WMMA) =================
  launch_gemm(EP_BN_RELU, x2,  lb1_w, lb1_b, scb1, shb1, zb1, GG, DB1,  DBIN, stream);
  launch_gemm(EP_BN_RELU, zb1, lb2_w, lb2_b, scb2, shb2, zb2, GG, DB2,  DB1,  stream);
  launch_gemm(EP_BN_RELU, zb2, lb3_w, lb3_b, scb3, shb3, zb3, GG, DB3,  DB2,  stream);
  launch_gemm(EP_BIAS_SIG, zb3, lb4_w, lb4_b, nullptr, nullptr, xb, GG, DBOUT, DB3, stream);

  // ================= combined head =================
  hipLaunchKernelGGL(k_comb1, dim3(cdiv(GG * DC, T)), dim3(T), 0, stream, xa, xb, lc1_w, lc1_b, cb);
  hipLaunchKernelGGL(k_comb2, dim3(cdiv(GG, T)), dim3(T), 0, stream, cb, lc2_w, lc2_b, out);
}
